// GNN_MLP_RNN_model_68564857914179
// MI455X (gfx1250) — matrix operand order repaired and checked by reading, not compiled, not run on hardware
//
#include <hip/hip_runtime.h>
#include <math.h>

// ---------------------------------------------------------------------------
// Problem constants: B=256, T=50, N=32, FIN=16, HG=64, HR=HFC=128,
// DS=DTG=32, OST=OCA=8.  M = B*T = 12800 rows.
// ---------------------------------------------------------------------------
#define MROWS 12800

typedef __attribute__((ext_vector_type(16))) _Float16 v16h;
typedef __attribute__((ext_vector_type(8)))  _Float16 v8h;
typedef __attribute__((ext_vector_type(8)))  float    v8f;

#define WMMA_F16(a, b, c) \
  __builtin_amdgcn_wmma_f32_16x16x32_f16(false, (a), false, (b), (short)0, (c), false, false)

// ---------------------------------------------------------------------------
// WMMA operand loaders (CDNA5 wave32 layouts, CDNA5 ISA).
// A 16x32 f16: lane l -> row = l&15, k = kb + (c&7) + ((c&8)<<1), kb=(l>>4)*8.
//   => per lane two contiguous 16B runs: [kb, kb+8) and [kb+16, kb+24).
// B 32x16 f16: lane l -> n = l&15, k same as A: kb + (c&7) + ((c&8)<<1).
//   => store B TRANSPOSED (T[n*ldK + k]) so each lane reads two 16B runs.
// D 16x16 f32: VGPR r -> m = (l>>4)*8 + r, n = l&15.
// ---------------------------------------------------------------------------
__device__ __forceinline__ v16h wmma_load_a_f16(const _Float16* p, int ld) {
  const int l = threadIdx.x & 31;
  const _Float16* q = p + (l & 15) * ld + ((l >> 4) << 3);
  const v8h lo = *(const v8h*)(q);
  const v8h hi = *(const v8h*)(q + 16);
  v16h a;
#pragma unroll
  for (int c = 0; c < 8; ++c) { a[c] = lo[c]; a[c + 8] = hi[c]; }
  return a;
}
__device__ __forceinline__ v16h wmma_load_a_f32(const float* p, int ld) {
  const int l = threadIdx.x & 31;
  const float* q = p + (l & 15) * ld + ((l >> 4) << 3);
  const float4 q0 = *(const float4*)(q);
  const float4 q1 = *(const float4*)(q + 4);
  const float4 q2 = *(const float4*)(q + 16);
  const float4 q3 = *(const float4*)(q + 20);
  v16h a;
  a[0] = (_Float16)q0.x;  a[1] = (_Float16)q0.y;
  a[2] = (_Float16)q0.z;  a[3] = (_Float16)q0.w;
  a[4] = (_Float16)q1.x;  a[5] = (_Float16)q1.y;
  a[6] = (_Float16)q1.z;  a[7] = (_Float16)q1.w;
  a[8] = (_Float16)q2.x;  a[9] = (_Float16)q2.y;
  a[10] = (_Float16)q2.z; a[11] = (_Float16)q2.w;
  a[12] = (_Float16)q3.x; a[13] = (_Float16)q3.y;
  a[14] = (_Float16)q3.z; a[15] = (_Float16)q3.w;
  return a;
}
// B from TRANSPOSED storage T[n*ldK + k]; p = tile origin (n-tile & k-step applied)
__device__ __forceinline__ v16h wmma_load_bt(const _Float16* p, int ldK) {
  const int l = threadIdx.x & 31;
  const _Float16* q = p + (l & 15) * ldK + ((l >> 4) << 3);
  const v8h lo = *(const v8h*)(q);
  const v8h hi = *(const v8h*)(q + 16);
  v16h b;
#pragma unroll
  for (int c = 0; c < 8; ++c) { b[c] = lo[c]; b[c + 8] = hi[c]; }
  return b;
}

__device__ __forceinline__ float sigmoidf_(float x) { return 1.0f / (1.0f + expf(-x)); }

// ---------------------------------------------------------------------------
// Converters: f32 row-major W(K,N) -> f16 transposed WT[n*K + k]
// ---------------------------------------------------------------------------
__global__ void convtr_f16_kernel(const float* __restrict__ W,
                                  _Float16* __restrict__ WT, int K, int N) {
  const int i = blockIdx.x * 256 + threadIdx.x;
  if (i < K * N) {
    const int k = i / N, n = i % N;
    WT[n * K + k] = (_Float16)W[i];
  }
}
// pack heads: WT[n*128 + k], n<8 from W_st(128x8), n in 8..15 from W_ca
__global__ void pack_heads_kernel(const float* __restrict__ Wst,
                                  const float* __restrict__ Wca,
                                  _Float16* __restrict__ WT) {
  const int i = blockIdx.x * 256 + threadIdx.x;
  if (i < 16 * 128) {
    const int n = i >> 7, k = i & 127;
    WT[i] = (_Float16)((n < 8) ? Wst[k * 8 + n] : Wca[k * 8 + (n - 8)]);
  }
}

// ---------------------------------------------------------------------------
// Kernel 1: GNN — one wave per (b,t).
//   h1 = relu(ln(adjS @ (feat @ W_g1) + b_g1)), adjS = (adj+I)/deg
//   out = relu(ln(adjS @ (h1 @ W_g3) + b_g3))[node 0]
// ---------------------------------------------------------------------------
__global__ __launch_bounds__(32) void gnn_kernel(
    const float* __restrict__ feat, const int* __restrict__ adj,
    const float* __restrict__ Wg1, const float* __restrict__ bg1,
    const float* __restrict__ Wg3, const float* __restrict__ bg3,
    const float* __restrict__ lng, const float* __restrict__ lnb,
    float* __restrict__ graph_seq) {
  __shared__ __align__(32) _Float16 sAdj[32 * 32];  // row-scaled adjacency (A)
  __shared__ __align__(32) _Float16 sX[32 * 64];    // activations, row-major (A)
  __shared__ __align__(32) _Float16 sPt[64 * 32];   // product, TRANSPOSED [n*32+m] (B)
  __shared__ __align__(32) _Float16 sWt[64 * 64];   // weights, TRANSPOSED [n*K+k] (B)
  __shared__ float sAcc[32 * 64];                   // f32 accum for LayerNorm

  const int idx = blockIdx.x;  // b*T + t
  const int l = threadIdx.x;   // 0..31

  // adjacency row l, +identity, pre-scaled by 1/deg
  const int* arow = adj + (size_t)idx * 32 * 32 + l * 32;
  float deg = 0.f;
#pragma unroll 8
  for (int k = 0; k < 32; ++k) deg += (float)arow[k] + (k == l ? 1.f : 0.f);
  const float inv = 1.f / deg;
#pragma unroll 8
  for (int k = 0; k < 32; ++k)
    sAdj[l * 32 + k] = (_Float16)(((float)arow[k] + (k == l ? 1.f : 0.f)) * inv);

  // feat row l (K=16 zero-padded to 32) -> sX (ld=32)
  const float* frow = feat + (size_t)idx * 32 * 16 + l * 16;
#pragma unroll
  for (int k = 0; k < 16; ++k) sX[l * 32 + k] = (_Float16)frow[k];
#pragma unroll
  for (int k = 16; k < 32; ++k) sX[l * 32 + k] = (_Float16)0.f;

  // W_g1 (16x64) -> transposed, K padded to 32: sWt[n*32 + k]
  for (int i = l; i < 64 * 32; i += 32) {
    const int n = i >> 5, k = i & 31;
    sWt[i] = (k < 16) ? (_Float16)Wg1[k * 64 + n] : (_Float16)0.f;
  }
  __syncthreads();

  // P1 = X @ W_g1 (32x64), store transposed into sPt[n*32 + m]
#pragma unroll
  for (int mt = 0; mt < 2; ++mt) {
    const v16h a = wmma_load_a_f16(sX + mt * 16 * 32, 32);
#pragma unroll
    for (int nt = 0; nt < 4; ++nt) {
      v8f c = {};
      c = WMMA_F16(a, wmma_load_bt(sWt + nt * 16 * 32, 32), c);
      const int n = nt * 16 + (l & 15);
      const int mb = mt * 16 + ((l >> 4) << 3);
#pragma unroll
      for (int r = 0; r < 8; ++r) sPt[n * 32 + (mb + r)] = (_Float16)c[r];
    }
  }
  __syncthreads();

  // agg1 = adjS @ P1 -> sAcc (f32)
#pragma unroll
  for (int mt = 0; mt < 2; ++mt) {
    const v16h a = wmma_load_a_f16(sAdj + mt * 16 * 32, 32);
#pragma unroll
    for (int nt = 0; nt < 4; ++nt) {
      v8f c = {};
      c = WMMA_F16(a, wmma_load_bt(sPt + nt * 16 * 32, 32), c);
      const int n = nt * 16 + (l & 15);
      const int mb = mt * 16 + ((l >> 4) << 3);
#pragma unroll
      for (int r = 0; r < 8; ++r) sAcc[(mb + r) * 64 + n] = c[r];
    }
  }
  __syncthreads();

  // bias + LN + ReLU (row l) -> sX as 32x64 f16 (row-major, A operand)
  {
    float s = 0.f;
    for (int c = 0; c < 64; ++c) {
      const float x = sAcc[l * 64 + c] + bg1[c];
      sAcc[l * 64 + c] = x;
      s += x;
    }
    const float m = s * (1.f / 64.f);
    float v = 0.f;
    for (int c = 0; c < 64; ++c) {
      const float d = sAcc[l * 64 + c] - m;
      v += d * d;
    }
    const float rstd = rsqrtf(v * (1.f / 64.f) + 1e-5f);
    for (int c = 0; c < 64; ++c) {
      const float y = (sAcc[l * 64 + c] - m) * rstd * lng[c] + lnb[c];
      sX[l * 64 + c] = (_Float16)fmaxf(y, 0.f);
    }
  }
  // W_g3 (64x64) -> transposed sWt[n*64 + k]
  for (int i = l; i < 64 * 64; i += 32) {
    const int n = i >> 6, k = i & 63;
    sWt[i] = (_Float16)Wg3[k * 64 + n];
  }
  __syncthreads();

  // P2 = h1 @ W_g3 (32x64, 2 K-steps), store transposed into sPt
#pragma unroll
  for (int mt = 0; mt < 2; ++mt) {
#pragma unroll
    for (int nt = 0; nt < 4; ++nt) {
      v8f c = {};
#pragma unroll
      for (int ks = 0; ks < 2; ++ks) {
        const v16h a = wmma_load_a_f16(sX + mt * 16 * 64 + ks * 32, 64);
        const v16h b = wmma_load_bt(sWt + nt * 16 * 64 + ks * 32, 64);
        c = WMMA_F16(a, b, c);
      }
      const int n = nt * 16 + (l & 15);
      const int mb = mt * 16 + ((l >> 4) << 3);
#pragma unroll
      for (int r = 0; r < 8; ++r) sPt[n * 32 + (mb + r)] = (_Float16)c[r];
    }
  }
  __syncthreads();

  // agg2 = adjS @ P2 -> sAcc.  Only row 0 is consumed downstream; still need
  // the m=0 tile only => mt = 0 suffices.
  {
    const v16h a = wmma_load_a_f16(sAdj, 32);
#pragma unroll
    for (int nt = 0; nt < 4; ++nt) {
      v8f c = {};
      c = WMMA_F16(a, wmma_load_bt(sPt + nt * 16 * 32, 32), c);
      const int n = nt * 16 + (l & 15);
      const int mb = (l >> 4) << 3;
#pragma unroll
      for (int r = 0; r < 8; ++r) sAcc[(mb + r) * 64 + n] = c[r];
    }
  }
  __syncthreads();

  // row 0: bias + LN + ReLU -> graph_seq[idx]
  if (l == 0) {
    float xbuf[64];
    float s = 0.f;
    for (int c = 0; c < 64; ++c) {
      xbuf[c] = sAcc[c] + bg3[c];
      s += xbuf[c];
    }
    const float m = s * (1.f / 64.f);
    float v = 0.f;
    for (int c = 0; c < 64; ++c) {
      const float d = xbuf[c] - m;
      v += d * d;
    }
    const float rstd = rsqrtf(v * (1.f / 64.f) + 1e-5f);
    for (int c = 0; c < 64; ++c) {
      const float y = (xbuf[c] - m) * rstd * lng[c] + lnb[c];
      graph_seq[(size_t)idx * 64 + c] = fmaxf(y, 0.f);
    }
  }
}

// ---------------------------------------------------------------------------
// Kernel 2: Y[m, n] = X(m,k) @ W(k,n) + bias[n], W given f16-TRANSPOSED.
// One wave per 16x16 tile.
// ---------------------------------------------------------------------------
__global__ __launch_bounds__(32) void gemm_bias_kernel(
    const float* __restrict__ X, const _Float16* __restrict__ WT,
    const float* __restrict__ bias, float* __restrict__ Y, int K, int ldY) {
  const int mt = blockIdx.x, nt = blockIdx.y;
  v8f c = {};
  for (int ks = 0; ks < K; ks += 32) {
    const v16h a = wmma_load_a_f32(X + (size_t)mt * 16 * K + ks, K);
    const v16h b = wmma_load_bt(WT + (size_t)nt * 16 * K + ks, K);
    c = WMMA_F16(a, b, c);
  }
  const int l = threadIdx.x & 31;
  const int n = nt * 16 + (l & 15);
  const int mb = mt * 16 + ((l >> 4) << 3);
#pragma unroll
  for (int r = 0; r < 8; ++r)
    Y[(size_t)(mb + r) * ldY + n] = c[r] + bias[n];
}

// ---------------------------------------------------------------------------
// Kernel 3: GRU, T=50 serial steps. Block = 16 batch rows, 4 waves.
// Wh resident in LDS as f16 TRANSPOSED [n*128 + k] (96 KB of 320 KB WGP LDS).
// ---------------------------------------------------------------------------
__global__ __launch_bounds__(128) void gru_kernel(
    const float* __restrict__ xp,   // (MROWS, 384), row = b*50+t
    const float* __restrict__ Wh,   // (128, 384) f32 row-major
    const float* __restrict__ bh,   // (384,)
    float* __restrict__ concat,     // (MROWS, 384), cols [col_off, col_off+128)
    int col_off) {
  extern __shared__ char smem[];
  _Float16* sWhT = (_Float16*)smem;                                   // 96 KB  [n*128+k]
  float*    sGh  = (float*)(smem + 128 * 384 * 2);                    // 24 KB
  float*    sH   = (float*)(smem + 128 * 384 * 2 + 16 * 384 * 4);     //  8 KB
  _Float16* sHh  = (_Float16*)(smem + 128 * 384 * 2 + 16 * 384 * 4 + 16 * 128 * 4); // 4 KB

  const int tid = threadIdx.x;
  const int wave = tid >> 5;
  const int btile = blockIdx.x;

  // coalesced global read, transposed scatter into LDS (one-time)
  for (int i = tid; i < 128 * 384; i += 128) {
    const int k = i / 384, n = i % 384;
    sWhT[n * 128 + k] = (_Float16)Wh[i];
  }
  for (int i = tid; i < 16 * 128; i += 128) {
    sH[i] = 0.f;
    sHh[i] = (_Float16)0.f;
  }
  __syncthreads();

  for (int t = 0; t < 50; ++t) {
    // gh = h @ Wh + bh : M=16, K=128 (4 steps), N=384 (24 tiles, 6/wave)
    const v16h a0 = wmma_load_a_f16(sHh + 0, 128);
    const v16h a1 = wmma_load_a_f16(sHh + 32, 128);
    const v16h a2 = wmma_load_a_f16(sHh + 64, 128);
    const v16h a3 = wmma_load_a_f16(sHh + 96, 128);
#pragma unroll
    for (int j = 0; j < 6; ++j) {
      const int nt = wave * 6 + j;
      const _Float16* wt = sWhT + nt * 16 * 128;
      v8f c = {};
      c = WMMA_F16(a0, wmma_load_bt(wt + 0, 128), c);
      c = WMMA_F16(a1, wmma_load_bt(wt + 32, 128), c);
      c = WMMA_F16(a2, wmma_load_bt(wt + 64, 128), c);
      c = WMMA_F16(a3, wmma_load_bt(wt + 96, 128), c);
      const int l = tid & 31;
      const int n = nt * 16 + (l & 15);
      const int mb = (l >> 4) << 3;
#pragma unroll
      for (int r = 0; r < 8; ++r) sGh[(mb + r) * 384 + n] = c[r] + bh[n];
    }
    __syncthreads();

    // fused gate math; each thread owns 16 of the 16x128 h elements
    for (int e = tid; e < 16 * 128; e += 128) {
      const int row = e >> 7, col = e & 127;
      const int gr = btile * 16 + row;
      const size_t xb = ((size_t)gr * 50 + t) * 384;
      const float ir = xp[xb + col];
      const float iz = xp[xb + 128 + col];
      const float ig = xp[xb + 256 + col];
      const float hr = sGh[row * 384 + col];
      const float hz = sGh[row * 384 + 128 + col];
      const float hg = sGh[row * 384 + 256 + col];
      const float rg = sigmoidf_(ir + hr);
      const float zg = sigmoidf_(iz + hz);
      const float ng = tanhf(ig + rg * hg);
      const float hn = (1.f - zg) * ng + zg * sH[e];
      sH[e] = hn;
      sHh[e] = (_Float16)hn;
      concat[xb + col_off + col] = hn;
    }
    // prefetch next timestep's xp rows (global_prefetch_b8)
    if (t < 49) {
      const int prow = tid >> 3, pseg = tid & 7;
      const size_t xb = ((size_t)(btile * 16 + prow) * 50 + t + 1) * 384;
      __builtin_prefetch(xp + xb + pseg * 48, 0, 0);
    }
    __syncthreads();
  }
}

// ---------------------------------------------------------------------------
// Kernel 4: h = relu(ln(concat@W_fc1 + b_fc1)); heads via packed 128x16 B.
// W_fc1 and heads given f16-TRANSPOSED. One wave per 16 rows.
// ---------------------------------------------------------------------------
__global__ __launch_bounds__(32) void fc_head_kernel(
    const float* __restrict__ concat, const _Float16* __restrict__ WfcT,
    const float* __restrict__ bfc, const float* __restrict__ lng,
    const float* __restrict__ lnb, const _Float16* __restrict__ WheadT,
    const float* __restrict__ bst, const float* __restrict__ bca,
    float* __restrict__ out, int Mrows) {
  __shared__ float sAcc[16 * 128];
  __shared__ __align__(32) _Float16 sHf[16 * 128];
  const int mt = blockIdx.x;
  const int l = threadIdx.x;
  const float* A = concat + (size_t)mt * 16 * 384;

#pragma unroll
  for (int nt = 0; nt < 8; ++nt) {
    v8f c = {};
    for (int ks = 0; ks < 12; ++ks) {
      const v16h a = wmma_load_a_f32(A + ks * 32, 384);
      const v16h b = wmma_load_bt(WfcT + (size_t)nt * 16 * 384 + ks * 32, 384);
      c = WMMA_F16(a, b, c);
    }
    const int n = nt * 16 + (l & 15);
    const int mb = (l >> 4) << 3;
#pragma unroll
    for (int r = 0; r < 8; ++r) sAcc[(mb + r) * 128 + n] = c[r] + bfc[n];
  }
  __syncthreads();

  if (l < 16) {
    float s = 0.f;
    for (int c = 0; c < 128; ++c) s += sAcc[l * 128 + c];
    const float m = s * (1.f / 128.f);
    float v = 0.f;
    for (int c = 0; c < 128; ++c) {
      const float d = sAcc[l * 128 + c] - m;
      v += d * d;
    }
    const float rstd = rsqrtf(v * (1.f / 128.f) + 1e-5f);
    for (int c = 0; c < 128; ++c) {
      const float y = (sAcc[l * 128 + c] - m) * rstd * lng[c] + lnb[c];
      sHf[l * 128 + c] = (_Float16)fmaxf(y, 0.f);
    }
  }
  __syncthreads();

  // heads: 16x128 @ 128x16 (st|ca packed), K=128 => 4 WMMA steps
  v8f c = {};
#pragma unroll
  for (int ks = 0; ks < 4; ++ks) {
    const v16h a = wmma_load_a_f16(sHf + ks * 32, 128);
    const v16h b = wmma_load_bt(WheadT + ks * 32, 128);
    c = WMMA_F16(a, b, c);
  }
  {
    const int n = l & 15;
    const int mb = (l >> 4) << 3;
#pragma unroll
    for (int r = 0; r < 8; ++r) {
      const int row = mt * 16 + mb + r;
      if (n < 8)
        out[(size_t)row * 8 + n] = c[r] + bst[n];
      else
        out[(size_t)Mrows * 8 + (size_t)row * 8 + (n - 8)] = c[r] + bca[n - 8];
    }
  }
}

// ---------------------------------------------------------------------------
extern "C" void kernel_launch(void* const* d_in, const int* in_sizes, int n_in,
                              void* d_out, int out_size, void* d_ws, size_t ws_size,
                              hipStream_t stream) {
  (void)in_sizes; (void)n_in; (void)out_size; (void)ws_size;
  const float* feat   = (const float*)d_in[0];
  const float* sensor = (const float*)d_in[1];
  const float* target = (const float*)d_in[2];
  const int*   adj    = (const int*)d_in[3];
  const float* Wg1 = (const float*)d_in[4];
  const float* bg1 = (const float*)d_in[5];
  const float* Wg3 = (const float*)d_in[6];
  const float* bg3 = (const float*)d_in[7];
  const float* lngg = (const float*)d_in[8];
  const float* lngb = (const float*)d_in[9];
  const float* WiG = (const float*)d_in[10];
  const float* WhG = (const float*)d_in[11];
  const float* biG = (const float*)d_in[12];
  const float* bhG = (const float*)d_in[13];
  const float* WiS = (const float*)d_in[14];
  const float* WhS = (const float*)d_in[15];
  const float* biS = (const float*)d_in[16];
  const float* bhS = (const float*)d_in[17];
  const float* WiT = (const float*)d_in[18];
  const float* WhT = (const float*)d_in[19];
  const float* biT = (const float*)d_in[20];
  const float* bhT = (const float*)d_in[21];
  const float* Wfc1 = (const float*)d_in[22];
  const float* bfc1 = (const float*)d_in[23];
  const float* lnfg = (const float*)d_in[24];
  const float* lnfb = (const float*)d_in[25];
  const float* Wst = (const float*)d_in[26];
  const float* bst = (const float*)d_in[27];
  const float* Wca = (const float*)d_in[28];
  const float* bca = (const float*)d_in[29];

  // f32 workspace
  float* ws = (float*)d_ws;
  float* graph_seq = ws;                        // 12800*64
  float* xpG = graph_seq + (size_t)MROWS * 64;  // 12800*384
  float* xpS = xpG + (size_t)MROWS * 384;
  float* xpT = xpS + (size_t)MROWS * 384;
  float* concat = xpT + (size_t)MROWS * 384;    // 12800*384
  // f16 transposed-weight workspace (after the f32 region; 32B-aligned)
  _Float16* hws = (_Float16*)(concat + (size_t)MROWS * 384);
  _Float16* WiGT  = hws;                // 64*384
  _Float16* WiST  = WiGT + 64 * 384;    // 32*384
  _Float16* WiTT  = WiST + 32 * 384;    // 32*384
  _Float16* WfcT  = WiTT + 32 * 384;    // 384*128
  _Float16* WheadT = WfcT + 384 * 128;  // 16*128
  float* out = (float*)d_out;

  // 0) one-time weight conversions (f32 row-major -> f16 transposed)
  convtr_f16_kernel<<<(64 * 384 + 255) / 256, 256, 0, stream>>>(WiG, WiGT, 64, 384);
  convtr_f16_kernel<<<(32 * 384 + 255) / 256, 256, 0, stream>>>(WiS, WiST, 32, 384);
  convtr_f16_kernel<<<(32 * 384 + 255) / 256, 256, 0, stream>>>(WiT, WiTT, 32, 384);
  convtr_f16_kernel<<<(384 * 128 + 255) / 256, 256, 0, stream>>>(Wfc1, WfcT, 384, 128);
  pack_heads_kernel<<<(16 * 128 + 255) / 256, 256, 0, stream>>>(Wst, Wca, WheadT);

  // 1) GNN
  gnn_kernel<<<MROWS, 32, 0, stream>>>(feat, adj, Wg1, bg1, Wg3, bg3, lngg, lngb, graph_seq);

  // 2) GRU input projections xp = x @ Wi + bi
  dim3 g2(MROWS / 16, 384 / 16);
  gemm_bias_kernel<<<g2, 32, 0, stream>>>(graph_seq, WiGT, biG, xpG, 64, 384);
  gemm_bias_kernel<<<g2, 32, 0, stream>>>(sensor,    WiST, biS, xpS, 32, 384);
  gemm_bias_kernel<<<g2, 32, 0, stream>>>(target,    WiTT, biT, xpT, 32, 384);

  // 3) GRU recurrences (Wh resident in LDS, f16 transposed)
  const size_t gru_lds = 128 * 384 * 2 + 16 * 384 * 4 + 16 * 128 * 4 + 16 * 128 * 2; // 135168 B
  gru_kernel<<<256 / 16, 128, gru_lds, stream>>>(xpG, WhG, bhG, concat, 0);
  gru_kernel<<<256 / 16, 128, gru_lds, stream>>>(xpS, WhS, bhS, concat, 128);
  gru_kernel<<<256 / 16, 128, gru_lds, stream>>>(xpT, WhT, bhT, concat, 256);

  // 4) FC + LN + ReLU + packed heads
  fc_head_kernel<<<MROWS / 16, 32, 0, stream>>>(concat, WfcT, bfc1, lnfg, lnfb,
                                                WheadT, bst, bca, out, MROWS);
}
